// AttentionLayer_46162308498182
// MI455X (gfx1250) — compile-verified, matrix operand order checked by reading, not run on hardware
//
#include <hip/hip_runtime.h>
#include <stdint.h>

// ---------------------------------------------------------------------------
// AttentionLayer for MI455X (gfx1250): bf16 WMMA everywhere.
//  - LDS-staged, double-buffered GEMMs using async load-to-LDS (ASYNCcnt)
//    issued via inline asm (clang builtin uses unspellable HIP LangAS params)
//  - fused attention with the 16x2048 score row-block + Q tile resident in LDS
// ---------------------------------------------------------------------------

typedef __attribute__((ext_vector_type(16))) __bf16 v16bf;
typedef __attribute__((ext_vector_type(8)))  float  v8f;

#define NX    1024
#define NH    1024
#define BATCH 8
#define SEQ   2048
#define MTOT  (BATCH * SEQ)   // 16384 total query rows across batches

static __device__ __constant__ float kScoreScale = 0.022097086912079608f; // 1/sqrt(2048)

// Flip to 0 to fall back to synchronous LDS staging (plain load+store).
#define USE_ASYNC_LDS_ASM 1

// Copy 16 bytes global -> LDS. Async path issues GLOBAL_LOAD_ASYNC_TO_LDS_B128
// (VGLOBAL op 98, tracked by ASYNCcnt; data bypasses VGPRs entirely).
// VDST = 32-bit LDS byte address (= low 32 bits of the generic shared pointer,
// per the ISA flat->LDS mapping); VADDR = 64-bit
// global address, GV mode ("off").
__device__ __forceinline__ void cp_b128(unsigned short* l, const unsigned short* g) {
#if USE_ASYNC_LDS_ASM
  unsigned int       loff  = (unsigned int)(uintptr_t)l;
  unsigned long long gaddr = (unsigned long long)(uintptr_t)g;
  asm volatile("global_load_async_to_lds_b128 %0, %1, off"
               :
               : "v"(loff), "v"(gaddr)
               : "memory");
#else
  *(uint4*)l = *(const uint4*)g;
#endif
}

template <int N>
__device__ __forceinline__ void wait_async() {
#if USE_ASYNC_LDS_ASM
  asm volatile("s_wait_asynccnt %0" :: "n"(N) : "memory");
#endif
}

// ---- bf16 helpers (raw-bits storage as unsigned short) ---------------------
__device__ __forceinline__ unsigned short f2bf(float x) {
  union { float f; uint32_t u; } c; c.f = x;
  uint32_t u = c.u;
  u += 0x7FFFu + ((u >> 16) & 1u);   // round-to-nearest-even
  return (unsigned short)(u >> 16);
}
__device__ __forceinline__ float bf2f(unsigned short h) {
  union { uint32_t u; float f; } c; c.u = ((uint32_t)h) << 16;
  return c.f;
}

// Load one 16x32 (A) or 32x16 (B) bf16 WMMA fragment: 16 bf16 per lane as
// two 16-byte chunks (VGPR0-3 = K-octet, VGPR4-7 = K-octet+16).
__device__ __forceinline__ v16bf load_frag16(const unsigned short* p) {
  union { uint4 q[2]; v16bf v; } f;
  f.q[0] = *(const uint4*)(p);
  f.q[1] = *(const uint4*)(p + 16);
  return f.v;
}

__device__ __forceinline__ v8f wmma_bf16(v16bf a, v16bf b, v8f c) {
  return __builtin_amdgcn_wmma_f32_16x16x32_bf16(
      /*neg_a=*/false, a, /*neg_b=*/false, b,
      /*c_mod=*/(short)0, c, /*reuse_a=*/false, /*reuse_b=*/false);
}

// ---- fp32 -> bf16 conversion ------------------------------------------------
__global__ void cvt_f32_to_bf16(const float* __restrict__ src,
                                unsigned short* __restrict__ dst, int n) {
  int i = blockIdx.x * blockDim.x + threadIdx.x;
  if (i < n) dst[i] = f2bf(src[i]);
}

// ---- LDS-staged, double-buffered bf16 WMMA GEMM: D = A[M,K] * B[N,K]^T -----
// B row-major [N][K] => B-matrix column n (= W row n) is contiguous in K.
// MODE 0: D bf16 row-major [M][N]
// MODE 1: D bf16 transposed [N][M] (packed b128 store per C fragment)
// MODE 2: D f32  row-major [M][N]
// Block = 256 threads (8 waves, 2Mx4N), block tile 32x256, K-step 32.
// Per K-slab: sA 32x32 (2 KB) + sB 256x32 (16 KB) staged via 5 async-copy
// instructions per wave, double buffered => s_wait_asynccnt 5 overlaps the
// prefetch of slab s+1 with the WMMAs of slab s.
template <int MODE>
__global__ __launch_bounds__(256)
void gemm_bf16(const unsigned short* __restrict__ A,
               const unsigned short* __restrict__ B,
               void* __restrict__ D, int M, int N, int K) {
  __shared__ __attribute__((aligned(16))) unsigned short sA[2][32 * 32];
  __shared__ __attribute__((aligned(16))) unsigned short sB[2][256 * 32];

  const int tid  = threadIdx.x;
  const int w    = tid >> 5;
  const int lane = tid & 31;
  const int hl   = lane >> 4;           // K-half select
  const int ln   = lane & 15;           // row (A) / col (B,C) within tile
  const int m_block = blockIdx.x * 32;
  const int n_block = blockIdx.y * 256;
  const int wm = (w >> 2) * 16;         // wave's local M base
  const int wn = (w & 3) * 64;          // wave's local N base

  // staging source pointers
  const unsigned short* gB = B + (size_t)(n_block + tid) * (size_t)K;  // col tid
  const int aIdx = w * 16 + ln;         // 0..127 (lanes 0-15 of each wave)
  const unsigned short* gA =
      A + (size_t)(m_block + (aIdx >> 2)) * (size_t)K + (aIdx & 3) * 8;

  // 4 B-copies (all lanes) + 1 A-copy (all lanes; 16-31 repeat 0-15) = 5 async issues per wave
  auto stage = [&](int buf, int k0) {
#pragma unroll
    for (int s2 = 0; s2 < 4; ++s2)
      cp_b128(&sB[buf][tid * 32 + s2 * 8], gB + k0 + s2 * 8);
    if (ln < 16)
      cp_b128(&sA[buf][(aIdx >> 2) * 32 + (aIdx & 3) * 8], gA + k0);
  };

  v8f acc[4];
#pragma unroll
  for (int i = 0; i < 4; ++i)
#pragma unroll
    for (int e = 0; e < 8; ++e) acc[i][e] = 0.0f;

  const int nslab = K / 32;
  stage(0, 0);
  for (int s = 0; s < nslab; ++s) {
    const int cur = s & 1;
    if (s + 1 < nslab) {
      stage(cur ^ 1, (s + 1) * 32);  // prefetch next slab
      wait_async<5>();               // wait only for the current slab
    } else {
      wait_async<0>();
    }
    __syncthreads();

    v16bf af = load_frag16(&sA[cur][(wm + ln) * 32 + hl * 8]);
#pragma unroll
    for (int i = 0; i < 4; ++i) {
      v16bf bf = load_frag16(&sB[cur][(wn + i * 16 + ln) * 32 + hl * 8]);
      acc[i] = wmma_bf16(af, bf, acc[i]);
    }
    __syncthreads();   // protect buffer `cur` before it is re-staged
  }

  const int m_base = m_block + wm;
#pragma unroll
  for (int i = 0; i < 4; ++i) {
    const int col = n_block + wn + i * 16 + ln;
    if (MODE == 0) {
      unsigned short* d = (unsigned short*)D;
#pragma unroll
      for (int r = 0; r < 8; ++r)
        d[(size_t)(m_base + hl * 8 + r) * (size_t)N + col] = f2bf(acc[i][r]);
    } else if (MODE == 1) {
      union { unsigned short h[8]; uint4 u; } pk;
#pragma unroll
      for (int r = 0; r < 8; ++r) pk.h[r] = f2bf(acc[i][r]);
      *(uint4*)((unsigned short*)D + (size_t)col * (size_t)M + m_base + hl * 8) = pk.u;
    } else {
      float* d = (float*)D;
#pragma unroll
      for (int r = 0; r < 8; ++r)
        d[(size_t)(m_base + hl * 8 + r) * (size_t)N + col] = acc[i][r];
    }
  }
}

// ---- fused attention: per (batch, 16-query-row tile) -----------------------
// Phase 0: stage Q tile [16,1024] bf16 into LDS (async copies)
// Phase 1: S[16,2048] = scale * Q_tile K^T   (bf16 in LDS, 512 WMMA / wave)
// Phase 2: row softmax of S in LDS (workgroup cooperative)
// Phase 3: O[16,1024] = P V  (A-frags from LDS, V read transposed, 512 WMMA)
__global__ __launch_bounds__(256)
void attn_fused(const unsigned short* __restrict__ qb,   // [MTOT][NX] bf16
                const unsigned short* __restrict__ kb,   // [MTOT][NX] bf16
                const unsigned short* __restrict__ vt,   // [NH][MTOT] bf16 (transposed)
                unsigned short* __restrict__ ob)         // [MTOT][NH] bf16
{
  __shared__ __attribute__((aligned(16))) unsigned short sS[16 * SEQ]; // 64 KB
  __shared__ __attribute__((aligned(16))) unsigned short sQ[16 * NX];  // 32 KB
  __shared__ float sRedA[256];
  __shared__ float sRedB[256];

  const int b   = blockIdx.y;
  const int mt  = blockIdx.x;
  const int tid = threadIdx.x;
  const int w    = tid >> 5;
  const int lane = tid & 31;
  const int hl   = lane >> 4;
  const int ln   = lane & 15;
  const size_t mrow0 = (size_t)b * SEQ + (size_t)mt * 16;

  // ---------------- Phase 0: stage Q tile (contiguous 32 KB) ---------------
  {
    const unsigned short* src = qb + mrow0 * (size_t)NX;
#pragma unroll
    for (int j = 0; j < 8; ++j) {
      const int c = j * 256 + tid;             // 2048 x 16B chunks
      cp_b128(&sQ[c * 8], src + (size_t)c * 8);
    }
  }
  wait_async<0>();
  __syncthreads();

  // ---------------- Phase 1: scores ----------------
  for (int jj = 0; jj < 16; ++jj) {
    const int jt = w * 16 + jj;                 // key tile 0..127 (wave's slice)
    v8f s;
#pragma unroll
    for (int e = 0; e < 8; ++e) s[e] = 0.0f;
    const unsigned short* aq = &sQ[ln * NX + hl * 8];
    const unsigned short* bk =
        kb + ((size_t)b * SEQ + (size_t)jt * 16 + ln) * (size_t)NX + hl * 8;
    for (int k0 = 0; k0 < NX; k0 += 32) {
      v16bf af = load_frag16(aq + k0);
      v16bf bf = load_frag16(bk + k0);
      s = wmma_bf16(af, bf, s);
    }
    const int col = jt * 16 + ln;
#pragma unroll
    for (int r = 0; r < 8; ++r)
      sS[(hl * 8 + r) * SEQ + col] = f2bf(s[r] * kScoreScale);
  }
  __syncthreads();

  // ---------------- Phase 2: softmax ----------------
  {
    const int row = tid >> 4;        // 16 rows x 16 segments of 128
    const int seg = tid & 15;
    unsigned short* p = &sS[row * SEQ + seg * 128];
    float lmax = -3.0e38f;
    for (int i = 0; i < 128; ++i) lmax = fmaxf(lmax, bf2f(p[i]));
    sRedA[tid] = lmax;
    __syncthreads();
    float rmax = -3.0e38f;
#pragma unroll
    for (int j = 0; j < 16; ++j) rmax = fmaxf(rmax, sRedA[row * 16 + j]);
    float lsum = 0.0f;
    for (int i = 0; i < 128; ++i) {
      float e = __expf(bf2f(p[i]) - rmax);
      lsum += e;
      p[i] = f2bf(e);
    }
    sRedB[tid] = lsum;
    __syncthreads();
    float rsum = 0.0f;
#pragma unroll
    for (int j = 0; j < 16; ++j) rsum += sRedB[row * 16 + j];
    const float inv = 1.0f / rsum;
    for (int i = 0; i < 128; ++i) p[i] = f2bf(bf2f(p[i]) * inv);
  }
  __syncthreads();

  // ---------------- Phase 3: O = P V ----------------
  v8f acc[8];
#pragma unroll
  for (int i = 0; i < 8; ++i)
#pragma unroll
    for (int e = 0; e < 8; ++e) acc[i][e] = 0.0f;

  const unsigned short* sA = &sS[ln * SEQ + hl * 8];   // A-frags from LDS
  for (int k0 = 0; k0 < SEQ; k0 += 32) {
    v16bf af = load_frag16(sA + k0);
#pragma unroll
    for (int i = 0; i < 8; ++i) {
      const int h = w * 128 + i * 16 + ln;             // wave owns 128 output cols
      const unsigned short* bp =
          vt + (size_t)h * (size_t)MTOT + (size_t)b * SEQ + k0 + hl * 8;
      v16bf bf = load_frag16(bp);
      acc[i] = wmma_bf16(af, bf, acc[i]);
    }
  }

#pragma unroll
  for (int i = 0; i < 8; ++i) {
    const int col = w * 128 + i * 16 + ln;
#pragma unroll
    for (int r = 0; r < 8; ++r)
      ob[(mrow0 + hl * 8 + r) * (size_t)NH + col] = f2bf(acc[i][r]);
  }
}

// ---------------------------------------------------------------------------
extern "C" void kernel_launch(void* const* d_in, const int* in_sizes, int n_in,
                              void* d_out, int out_size, void* d_ws, size_t ws_size,
                              hipStream_t stream) {
  (void)in_sizes; (void)n_in; (void)out_size; (void)ws_size;

  const float* x  = (const float*)d_in[0];   // [8,2048,1024]
  const float* Wq = (const float*)d_in[1];   // [1024,1024]
  const float* Wk = (const float*)d_in[2];
  const float* Wv = (const float*)d_in[3];
  const float* Wo = (const float*)d_in[4];
  float* out = (float*)d_out;                // [8,2048,1024] f32

  // Workspace layout (bf16 shorts), total ~168 MB
  unsigned short* ws  = (unsigned short*)d_ws;
  const size_t NXELT = (size_t)MTOT * NX;      // 16,777,216
  const size_t WELT  = (size_t)NH * NX;        // 1,048,576
  unsigned short* xb   = ws;
  unsigned short* wqb  = xb  + NXELT;
  unsigned short* wkb  = wqb + WELT;
  unsigned short* wvb  = wkb + WELT;
  unsigned short* wob  = wvb + WELT;
  unsigned short* qb   = wob + WELT;
  unsigned short* kbuf = qb  + NXELT;
  unsigned short* vtb  = kbuf + NXELT;         // [NH][MTOT]
  unsigned short* obuf = vtb + NXELT;          // pre-projection attn output

  // 1) fp32 -> bf16 converts
  cvt_f32_to_bf16<<<(int)((NXELT + 255) / 256), 256, 0, stream>>>(x, xb, (int)NXELT);
  cvt_f32_to_bf16<<<(int)((WELT + 255) / 256), 256, 0, stream>>>(Wq, wqb, (int)WELT);
  cvt_f32_to_bf16<<<(int)((WELT + 255) / 256), 256, 0, stream>>>(Wk, wkb, (int)WELT);
  cvt_f32_to_bf16<<<(int)((WELT + 255) / 256), 256, 0, stream>>>(Wv, wvb, (int)WELT);
  cvt_f32_to_bf16<<<(int)((WELT + 255) / 256), 256, 0, stream>>>(Wo, wob, (int)WELT);

  // 2) QKV projections (V stored transposed for the PV GEMM's B-layout)
  dim3 gg(MTOT / 32, NH / 256);   // (512, 4)
  gemm_bf16<0><<<gg, 256, 0, stream>>>(xb, wqb, qb,   MTOT, NH, NX);
  gemm_bf16<0><<<gg, 256, 0, stream>>>(xb, wkb, kbuf, MTOT, NH, NX);
  gemm_bf16<1><<<gg, 256, 0, stream>>>(xb, wvb, vtb,  MTOT, NH, NX);

  // 3) fused attention: scores + softmax + PV, scores + Q tile resident in LDS
  attn_fused<<<dim3(SEQ / 16, BATCH), 256, 0, stream>>>(qb, kbuf, vtb, obuf);

  // 4) output projection, fp32 store to d_out
  gemm_bf16<2><<<gg, 256, 0, stream>>>(obuf, wob, out, MTOT, NH, NH);
}
